// Attention_1322849927460
// MI455X (gfx1250) — matrix operand order repaired and checked by reading, not compiled, not run on hardware
//
#include <hip/hip_runtime.h>
#include <math.h>

// ---------------------------------------------------------------------------
// Types for CDNA5 WMMA (wave32, 16x16x32 f16 -> f32)
// ---------------------------------------------------------------------------
typedef __attribute__((ext_vector_type(16))) _Float16 v16h;
typedef __attribute__((ext_vector_type(8)))  _Float16 v8h;
typedef __attribute__((ext_vector_type(8)))  float    v8f;

#define BATCH 2
#define MTOK  2048          // l*n tokens per batch
#define BM    4096          // BATCH*MTOK rows
#define DIM   256
#define HEADS 8
#define DH    64
#define INNER 512
#define NQKV  1536
#define VEXT  80            // DH + 3 xyz cols, padded to 16

static __device__ __forceinline__ v8f wmma_f16(v16h a, v16h b, v8f c) {
  return __builtin_amdgcn_wmma_f32_16x16x32_f16(false, a, false, b, (short)0, c,
                                                false, false);
}

// A-fragment (16x32 f16, M x K): lane L holds row m=L&15; halfs 0..7 = K 8*hi..8*hi+7,
// halfs 8..15 = K 16+8*hi..16+8*hi+7  (ISA 7.12.2).  tile = row-major, ld in halfs.
static __device__ __forceinline__ v16h load_a_frag(const _Float16* tile, int ld, int lane) {
  int m = lane & 15, hi = lane >> 4;
  const v8h* p0 = (const v8h*)(tile + (size_t)m * ld + 8 * hi);
  const v8h* p1 = (const v8h*)(tile + (size_t)m * ld + 16 + 8 * hi);
  v8h x = *p0, y = *p1;
  v16h a;
#pragma unroll
  for (int i = 0; i < 8; ++i) { a[i] = x[i]; a[i + 8] = y[i]; }
  return a;
}

// B-fragment (32x16 f16, K x N): lane L holds col n=L&15; halfs 0..7 = K 8*hi..8*hi+7,
// halfs 8..15 = K 16+8*hi..16+8*hi+7.
// "tile" is stored N-major: row n contiguous over K (ld = stride between n rows, halfs).
static __device__ __forceinline__ v16h load_b_frag(const _Float16* tile, int ld, int lane) {
  int n = lane & 15, hi = lane >> 4;
  const v8h* p = (const v8h*)(tile + (size_t)n * ld + 8 * hi);
  v8h x = p[0], y = p[2];
  v16h b;
#pragma unroll
  for (int i = 0; i < 8; ++i) { b[i] = x[i]; b[i + 8] = y[i]; }
  return b;
}

// reductions across the 16-lane half (rows of the C layout share L>>4)
static __device__ __forceinline__ float hmax16(float v) {
  v = fmaxf(v, __shfl_xor(v, 1, 32));
  v = fmaxf(v, __shfl_xor(v, 2, 32));
  v = fmaxf(v, __shfl_xor(v, 4, 32));
  v = fmaxf(v, __shfl_xor(v, 8, 32));
  return v;
}
static __device__ __forceinline__ float hsum16(float v) {
  v += __shfl_xor(v, 1, 32);
  v += __shfl_xor(v, 2, 32);
  v += __shfl_xor(v, 4, 32);
  v += __shfl_xor(v, 8, 32);
  return v;
}
static __device__ __forceinline__ float hsum32(float v) {
  v += __shfl_xor(v, 1, 32);
  v += __shfl_xor(v, 2, 32);
  v += __shfl_xor(v, 4, 32);
  v += __shfl_xor(v, 8, 32);
  v += __shfl_xor(v, 16, 32);
  return v;
}
static __device__ __forceinline__ float gelu_f(float x) {
  return 0.5f * x * (1.0f + erff(x * 0.70710678118654752f));
}

// ---------------------------------------------------------------------------
// 1) Transpose+convert weights to f16 (B-fragment friendly: n rows contiguous in k)
// ---------------------------------------------------------------------------
__global__ void prep_weights_k(const float* __restrict__ w_qkv,
                               const float* __restrict__ out_w,
                               _Float16* __restrict__ wqkvT,
                               _Float16* __restrict__ outwT) {
  int idx = blockIdx.x * blockDim.x + threadIdx.x;
  const int NW1 = NQKV * DIM;           // 1536*256
  if (idx < NW1) {
    int nn = idx >> 8, kk = idx & 255;                    // wqkvT[n][k]
    wqkvT[idx] = (_Float16)w_qkv[(size_t)kk * NQKV + nn]; // w_qkv is (256,1536)
  } else {
    int j = idx - NW1;
    if (j < DIM * INNER) {
      int nn = j >> 9, kk = j & 511;                      // outwT[n][k], n<256,k<512
      outwT[j] = (_Float16)out_w[(size_t)kk * DIM + nn];  // out_w is (512,256)
    }
  }
}

// ---------------------------------------------------------------------------
// 2) LayerNorm -> xn16 (f16), one wave per row of 256
// ---------------------------------------------------------------------------
__global__ void layernorm_k(const float* __restrict__ x, const float* __restrict__ g,
                            const float* __restrict__ bta, _Float16* __restrict__ xn16) {
  int lane = threadIdx.x & 31, wv = threadIdx.x >> 5;
  int row = blockIdx.x * 8 + wv;                     // 4096 rows, grid=512 x 256
  const float* xr = x + (size_t)row * DIM;
  float v[8], s = 0.f;
#pragma unroll
  for (int i = 0; i < 8; ++i) { v[i] = xr[lane * 8 + i]; s += v[i]; }
  float mu = hsum32(s) * (1.0f / DIM);
  float q = 0.f;
#pragma unroll
  for (int i = 0; i < 8; ++i) { float d = v[i] - mu; q += d * d; }
  float rstd = rsqrtf(hsum32(q) * (1.0f / DIM) + 1e-5f);
#pragma unroll
  for (int i = 0; i < 8; ++i) {
    int c = lane * 8 + i;
    xn16[(size_t)row * DIM + c] = (_Float16)((v[i] - mu) * rstd * g[c] + bta[c]);
  }
}

// ---------------------------------------------------------------------------
// 3) QKV GEMM: (4096x256)@(256x1536), scatter into head-major q/k and transposed v
// ---------------------------------------------------------------------------
__global__ void qkv_gemm_k(const _Float16* __restrict__ xn16,
                           const _Float16* __restrict__ wqkvT,
                           _Float16* __restrict__ q16, _Float16* __restrict__ k16,
                           _Float16* __restrict__ vt16) {
  int lane = threadIdx.x & 31, wv = threadIdx.x >> 5;
  int tile = blockIdx.x * 4 + wv;                    // 24576 tiles, grid=6144 x 128
  int mt = tile / 96, nt = tile % 96;
  v8f acc = {};
#pragma unroll
  for (int kk = 0; kk < 8; ++kk) {                   // K = 256 = 8*32
    v16h a = load_a_frag(xn16 + (size_t)mt * 16 * DIM + kk * 32, DIM, lane);
    v16h b = load_b_frag(wqkvT + (size_t)nt * 16 * DIM + kk * 32, DIM, lane);
    acc = wmma_f16(a, b, acc);
  }
  int n = lane & 15, hi = lane >> 4;
#pragma unroll
  for (int r = 0; r < 8; ++r) {
    int row = mt * 16 + r + 8 * hi;
    int col = nt * 16 + n;
    float val = acc[r];
    int sect = col >> 9, within = col & 511;
    int head = within >> 6, c = within & 63;
    int bidx = row >> 11, mm = row & 2047;
    int bh = (bidx << 3) + head;
    if (sect == 0) {
      q16[(((size_t)bh << 11) + mm) * DH + c] = (_Float16)(val * 0.125f); // fold 1/sqrt(64)
    } else if (sect == 1) {
      k16[(((size_t)bh << 11) + mm) * DH + c] = (_Float16)val;
    } else {
      vt16[((size_t)bh * VEXT + c) * MTOK + mm] = (_Float16)val;          // transposed V
    }
  }
}

// ---------------------------------------------------------------------------
// 4) Fill vt rows 64..79 with xyz columns (64..66) and zero padding (67..79)
// ---------------------------------------------------------------------------
__global__ void xyz_fill_k(const float* __restrict__ xyzs, _Float16* __restrict__ vt16) {
  int idx = blockIdx.x * blockDim.x + threadIdx.x;   // 16*16*2048
  int m = idx & 2047, c = (idx >> 11) & 15, bh = idx >> 15;
  int b = bh >> 3;
  float val = (c < 3) ? xyzs[((size_t)b * MTOK + m) * 3 + c] : 0.0f;
  vt16[((size_t)bh * VEXT + 64 + c) * MTOK + m] = (_Float16)val;
}

// ---------------------------------------------------------------------------
// 5) Flash attention: one wave per (bh, 16-row i-tile). V extended with xyz, so
//    acc tile 4 yields attn@xyz; delta_full = attn@xyz - xyz_i (softmax sums to 1).
// ---------------------------------------------------------------------------
__global__ void flash_attn_k(const _Float16* __restrict__ q16,
                             const _Float16* __restrict__ k16,
                             const _Float16* __restrict__ vt16,
                             const float* __restrict__ xyzs,
                             float* __restrict__ out_attn,   // (b, m, 512) f32
                             float* __restrict__ delta) {    // (bh, m, 3)  f32
  __shared__ __align__(16) _Float16 p_lds[16][32];
  int lane = threadIdx.x;                            // blockDim = 32
  int bid = blockIdx.x;                              // 16 bh * 128 i-tiles
  int bh = bid >> 7, it = bid & 127;
  int i0 = it * 16;
  int b = bh >> 3, h = bh & 7;
  int n = lane & 15, hi = lane >> 4;

  const _Float16* qb = q16 + ((size_t)bh * MTOK + i0) * DH;
  v16h a0 = load_a_frag(qb, DH, lane);               // K 0..31 of dh
  v16h a1 = load_a_frag(qb + 32, DH, lane);          // K 32..63

  v8f acc[5];
  float mst[8], lst[8];
#pragma unroll
  for (int t = 0; t < 5; ++t) { v8f z = {}; acc[t] = z; }
#pragma unroll
  for (int r = 0; r < 8; ++r) { mst[r] = -3.0e38f; lst[r] = 0.f; }

  for (int j0 = 0; j0 < MTOK; j0 += 32) {
    const _Float16* kb = k16 + ((size_t)bh * MTOK + j0) * DH;
    // S = Q @ K^T  (two 16x16 C tiles: keys j0..+15, j0+16..+31)
    v8f s0 = {}, s1 = {};
    s0 = wmma_f16(a0, load_b_frag(kb, DH, lane), s0);
    s0 = wmma_f16(a1, load_b_frag(kb + 32, DH, lane), s0);
    const _Float16* kb2 = kb + 16 * DH;
    s1 = wmma_f16(a0, load_b_frag(kb2, DH, lane), s1);
    s1 = wmma_f16(a1, load_b_frag(kb2 + 32, DH, lane), s1);

    // online softmax per row (row r+8*hi lives in VGPR r across the 16-lane half)
#pragma unroll
    for (int r = 0; r < 8; ++r) {
      float rmax = hmax16(fmaxf(s0[r], s1[r]));
      float nm = fmaxf(mst[r], rmax);
      float al = __expf(mst[r] - nm);
      float p0 = __expf(s0[r] - nm);
      float p1 = __expf(s1[r] - nm);
      lst[r] = lst[r] * al + hsum16(p0 + p1);
      mst[r] = nm;
#pragma unroll
      for (int t = 0; t < 5; ++t) acc[t][r] *= al;
      p_lds[r + 8 * hi][n] = (_Float16)p0;           // C layout -> LDS row-major
      p_lds[r + 8 * hi][16 + n] = (_Float16)p1;
    }
    __syncthreads();
    v16h pa = load_a_frag(&p_lds[0][0], 32, lane);   // reload as A-fragment
#pragma unroll
    for (int t = 0; t < 5; ++t) {
      v16h bv = load_b_frag(vt16 + ((size_t)bh * VEXT + 16 * t) * MTOK + j0, MTOK, lane);
      acc[t] = wmma_f16(pa, bv, acc[t]);
    }
    __syncthreads();
  }

  // epilogue: normalize; tiles 0..3 -> out_attn, tile 4 cols 0..2 -> delta
#pragma unroll
  for (int r = 0; r < 8; ++r) {
    float inv = 1.0f / lst[r];
    int mm = i0 + r + 8 * hi;
    size_t obase = ((size_t)b * MTOK + mm) * INNER + h * DH;
#pragma unroll
    for (int t = 0; t < 4; ++t)
      out_attn[obase + t * 16 + n] = acc[t][r] * inv;
    if (n < 3)
      delta[((size_t)bh * MTOK + mm) * 3 + n] =
          acc[4][r] * inv - xyzs[((size_t)b * MTOK + mm) * 3 + n];
  }
}

// ---------------------------------------------------------------------------
// 6) Spatial MLP per (bh,row): hid = gelu(delta@W1+b1) [512]; bias = hid@W2+b2 [64];
//    add to attention output, store f16 for the final GEMM.
// ---------------------------------------------------------------------------
__global__ void spatial_k(const float* __restrict__ delta,
                          const float* __restrict__ sp_w1, const float* __restrict__ sp_b1,
                          const float* __restrict__ sp_w2, const float* __restrict__ sp_b2,
                          const float* __restrict__ out_attn,
                          _Float16* __restrict__ attnout16) {
  __shared__ float hid[8][512];
  int lane = threadIdx.x & 31, wv = threadIdx.x >> 5;
  int row = blockIdx.x * 8 + wv;                     // bh*2048+mm, grid=4096 x 256
  int bh = row >> 11, mm = row & 2047;
  int b = bh >> 3, h = bh & 7;
  const float* dr = delta + (size_t)row * 3;
  float d0 = dr[0], d1 = dr[1], d2 = dr[2];
#pragma unroll
  for (int ii = 0; ii < 16; ++ii) {
    int u = ii * 32 + lane;
    float t = d0 * sp_w1[u] + d1 * sp_w1[512 + u] + d2 * sp_w1[1024 + u] + sp_b1[u];
    hid[wv][u] = gelu_f(t);
  }
  __syncthreads();
  int c0 = lane * 2, c1 = c0 + 1;
  float a0 = sp_b2[c0], a1 = sp_b2[c1];
  for (int u = 0; u < 512; ++u) {
    float hv = hid[wv][u];
    a0 += hv * sp_w2[(size_t)u * DH + c0];
    a1 += hv * sp_w2[(size_t)u * DH + c1];
  }
  size_t base = ((size_t)b * MTOK + mm) * INNER + h * DH;
  attnout16[base + c0] = (_Float16)(out_attn[base + c0] + a0);
  attnout16[base + c1] = (_Float16)(out_attn[base + c1] + a1);
}

// ---------------------------------------------------------------------------
// 7) Output GEMM: gelu((4096x512)@(512x256) + out_b) + features -> d_out (f32)
// ---------------------------------------------------------------------------
__global__ void out_gemm_k(const _Float16* __restrict__ a16,
                           const _Float16* __restrict__ outwT,
                           const float* __restrict__ out_b,
                           const float* __restrict__ features,
                           float* __restrict__ dout) {
  int lane = threadIdx.x & 31, wv = threadIdx.x >> 5;
  int tile = blockIdx.x * 4 + wv;                    // 256*16 tiles, grid=1024 x 128
  int mt = tile >> 4, nt = tile & 15;
  v8f acc = {};
#pragma unroll
  for (int kk = 0; kk < 16; ++kk) {                  // K = 512
    v16h a = load_a_frag(a16 + (size_t)mt * 16 * INNER + kk * 32, INNER, lane);
    v16h b = load_b_frag(outwT + (size_t)nt * 16 * INNER + kk * 32, INNER, lane);
    acc = wmma_f16(a, b, acc);
  }
  int n = lane & 15, hi = lane >> 4;
#pragma unroll
  for (int r = 0; r < 8; ++r) {
    int row = mt * 16 + r + 8 * hi;
    int col = nt * 16 + n;
    float v = gelu_f(acc[r] + out_b[col]) + features[(size_t)row * DIM + col];
    dout[(size_t)row * DIM + col] = v;
  }
}

// ---------------------------------------------------------------------------
extern "C" void kernel_launch(void* const* d_in, const int* in_sizes, int n_in,
                              void* d_out, int out_size, void* d_ws, size_t ws_size,
                              hipStream_t stream) {
  const float* xyzs     = (const float*)d_in[0];
  const float* features = (const float*)d_in[1];
  const float* ln_g     = (const float*)d_in[2];
  const float* ln_b     = (const float*)d_in[3];
  const float* w_qkv    = (const float*)d_in[4];
  const float* sp_w1    = (const float*)d_in[5];
  const float* sp_b1    = (const float*)d_in[6];
  const float* sp_w2    = (const float*)d_in[7];
  const float* sp_b2    = (const float*)d_in[8];
  const float* out_w    = (const float*)d_in[9];
  const float* out_b    = (const float*)d_in[10];
  float* dout = (float*)d_out;

  // workspace carve-up (all offsets 256B aligned)
  size_t off = 0;
  auto carve = [&](size_t bytes) { size_t o = off; off = (off + bytes + 255) & ~(size_t)255; return o; };
  char* ws = (char*)d_ws;
  _Float16* xn16    = (_Float16*)(ws + carve((size_t)BM * DIM * 2));
  _Float16* wqkvT   = (_Float16*)(ws + carve((size_t)NQKV * DIM * 2));
  _Float16* outwT   = (_Float16*)(ws + carve((size_t)DIM * INNER * 2));
  _Float16* q16     = (_Float16*)(ws + carve((size_t)16 * MTOK * DH * 2));
  _Float16* k16     = (_Float16*)(ws + carve((size_t)16 * MTOK * DH * 2));
  _Float16* vt16    = (_Float16*)(ws + carve((size_t)16 * VEXT * MTOK * 2));
  float*    outattn = (float*)   (ws + carve((size_t)BM * INNER * 4));
  float*    delta   = (float*)   (ws + carve((size_t)16 * MTOK * 3 * 4));
  _Float16* ao16    = (_Float16*)(ws + carve((size_t)BM * INNER * 2));
  (void)in_sizes; (void)n_in; (void)out_size; (void)ws_size;

  prep_weights_k<<<2048, 256, 0, stream>>>(w_qkv, out_w, wqkvT, outwT);
  layernorm_k<<<512, 256, 0, stream>>>(features, ln_g, ln_b, xn16);
  qkv_gemm_k<<<6144, 128, 0, stream>>>(xn16, wqkvT, q16, k16, vt16);
  xyz_fill_k<<<2048, 256, 0, stream>>>(xyzs, vt16);
  flash_attn_k<<<2048, 32, 0, stream>>>(q16, k16, vt16, xyzs, outattn, delta);
  spatial_k<<<4096, 256, 0, stream>>>(delta, sp_w1, sp_b1, sp_w2, sp_b2, outattn, ao16);
  out_gemm_k<<<1024, 128, 0, stream>>>(ao16, outwT, out_b, features, dout);
}
